// Model_39676907888050
// MI455X (gfx1250) — matrix operand order repaired and checked by reading, not compiled, not run on hardware
//
#include <hip/hip_runtime.h>

typedef __attribute__((ext_vector_type(16))) _Float16 v16h;
typedef __attribute__((ext_vector_type(8)))  _Float16 v8h;
typedef __attribute__((ext_vector_type(8)))  float    v8f;

#define SEQ   2048
#define DIM   64
#define NBATCH 8

// DPP16 ROW_XMASK cross-lane move within each row of 16 lanes (pure VALU).
template <int CTRL>
__device__ __forceinline__ float dpp_xor_f32(float x) {
    return __builtin_bit_cast(
        float, __builtin_amdgcn_update_dpp(
                   0, __builtin_bit_cast(int, x), CTRL,
                   /*row_mask=*/0xf, /*bank_mask=*/0xf, /*bound_ctrl=*/true));
}

__device__ __forceinline__ float rowmax16(float x) {
    x = fmaxf(x, dpp_xor_f32<0x161>(x));   // xor 1
    x = fmaxf(x, dpp_xor_f32<0x162>(x));   // xor 2
    x = fmaxf(x, dpp_xor_f32<0x164>(x));   // xor 4
    x = fmaxf(x, dpp_xor_f32<0x168>(x));   // xor 8
    return x;
}

__device__ __forceinline__ float rowsum16(float x) {
    x += dpp_xor_f32<0x161>(x);
    x += dpp_xor_f32<0x162>(x);
    x += dpp_xor_f32<0x164>(x);
    x += dpp_xor_f32<0x168>(x);
    return x;
}

// ---------------------------------------------------------------------------
// Kernel 1: linear projection  Y = X @ W + b  (fp32 in, f16 out)
// One block = 64 rows of one (matrix, batch). W (16KB) cached in LDS.
// transpose==1 writes Y^T as [batch][col][row] (V path).
// scale_ptr!=nullptr: multiply result by 1/scale_ptr[0] (Q path — folds the
// attention scale out of the flash kernel's hot loop).
// ---------------------------------------------------------------------------
__global__ __launch_bounds__(256)
void proj_kernel(const float* __restrict__ X, const float* __restrict__ W,
                 const float* __restrict__ bias, _Float16* __restrict__ Y,
                 int transpose, const float* __restrict__ scale_ptr) {
    __shared__ float Wl[DIM * DIM];
    __shared__ float bl[DIM];
    const int t = threadIdx.x;
    for (int i = t; i < DIM * DIM; i += 256) Wl[i] = W[i];
    if (t < DIM) bl[t] = bias[t];
    __syncthreads();

    const float sc = scale_ptr ? (1.0f / scale_ptr[0]) : 1.0f;

    const int bidx  = blockIdx.x;          // batch*32 + tile
    const int batch = bidx >> 5;
    const int tile  = bidx & 31;
    const int col   = t & 63;
    const int rbase = t >> 6;              // 0..3

    const float* xb = X + ((size_t)batch * SEQ + (size_t)tile * 64) * DIM;
    for (int rr = rbase; rr < 64; rr += 4) {
        const float* xr = xb + (size_t)rr * DIM;
        float acc = bl[col];
#pragma unroll
        for (int k2 = 0; k2 < DIM; ++k2)
            acc = fmaf(xr[k2], Wl[k2 * DIM + col], acc);
        acc *= sc;
        const int row = tile * 64 + rr;
        if (transpose)
            Y[(size_t)batch * DIM * SEQ + (size_t)col * SEQ + row] = (_Float16)acc;
        else
            Y[((size_t)batch * SEQ + row) * DIM + col] = (_Float16)acc;
    }
}

// ---------------------------------------------------------------------------
// Kernel 2: flash attention, one wave32 per 16-row Q tile, 64-wide KV chunks.
//   Q : f16 [b][s][64] pre-scaled  (A-operand: 2x16B loads/lane)
//   K : f16 [b][s][64]             (B-operand: 32B contiguous loads/lane)
//   Vt: f16 [b][64][s]             (B-operand: 32B contiguous loads/lane)
// 16 WMMAs per iteration; online softmax bookkeeping once per 64 columns.
// l kept lane-local and reduced once at the end. P relayout via 2KB LDS,
// tile-major [tile][m][16].
// ---------------------------------------------------------------------------
__global__ __launch_bounds__(32)
void flash_attn(const _Float16* __restrict__ Q,
                const _Float16* __restrict__ K,
                const _Float16* __restrict__ Vt,
                const float* __restrict__ dropout_p,
                float* __restrict__ out) {
    __shared__ __align__(32) _Float16 Pld[4 * 16 * 16];   // [tile][m][16]

    const int lane = threadIdx.x;   // 0..31
    const int hi   = lane >> 4;     // half-wave id
    const int lo   = lane & 15;

    const int bid   = blockIdx.x;   // b*128 + qtile
    const int b     = bid >> 7;
    const int qrow0 = (bid & 127) * 16;

    const float dropscale = 1.0f / (1.0f - dropout_p[0]);

    // ---- Q A-operands for contraction chunks d=0..31 and d=32..63 ----
    const _Float16* qbase = Q + ((size_t)b * SEQ + qrow0 + lo) * DIM;
    v8h q00 = *(const v8h*)(qbase + hi * 16);
    v8h q01 = *(const v8h*)(qbase + 8 + hi * 16);
    v8h q10 = *(const v8h*)(qbase + 32 + hi * 16);
    v8h q11 = *(const v8h*)(qbase + 40 + hi * 16);
    v16h qa0, qa1;
#pragma unroll
    for (int i = 0; i < 8; ++i) {
        qa0[i] = q00[i]; qa0[i + 8] = q01[i];
        qa1[i] = q10[i]; qa1[i + 8] = q11[i];
    }

    const _Float16* kbase = K  + (size_t)b * SEQ * DIM;
    const _Float16* vtb   = Vt + (size_t)b * DIM * SEQ;

    v8f acc[4] = {v8f{}, v8f{}, v8f{}, v8f{}};
    float mrun[8], lpart[8];
#pragma unroll
    for (int r = 0; r < 8; ++r) { mrun[r] = -3.0e38f; lpart[r] = 0.0f; }

    for (int kv = 0; kv < SEQ; kv += 64) {
        // ---- scores: four 16x16 tiles (cols kv + 16*t2 + 0..15) ----
        v8f s[4];
#pragma unroll
        for (int t2 = 0; t2 < 4; ++t2) {
            const _Float16* kp = kbase + (size_t)(kv + t2 * 16 + lo) * DIM + hi * 16;
            v16h ka = *(const v16h*)(kp);        // contraction d = hi*16 + 0..15
            v16h kc = *(const v16h*)(kp + 32);   // contraction d = 32 + ...
            v8f sz = {};
            sz = __builtin_amdgcn_wmma_f32_16x16x32_f16(false, qa0, false, ka, (short)0, sz, false, false);
            sz = __builtin_amdgcn_wmma_f32_16x16x32_f16(false, qa1, false, kc, (short)0, sz, false, false);
            s[t2] = sz;
        }

        // ---- online softmax (row = r + 8*hi, spread over 16 lanes) ----
        float p0[8], p1[8], p2[8], p3[8], corr[8];
#pragma unroll
        for (int r = 0; r < 8; ++r) {
            float cm = fmaxf(fmaxf(s[0][r], s[1][r]), fmaxf(s[2][r], s[3][r]));
            cm = rowmax16(cm);                         // DPP, no LDS
            float newm = fmaxf(mrun[r], cm);
            float cf   = __expf(mrun[r] - newm);
            mrun[r]    = newm;
            float e0 = __expf(s[0][r] - newm);
            float e1 = __expf(s[1][r] - newm);
            float e2 = __expf(s[2][r] - newm);
            float e3 = __expf(s[3][r] - newm);
            lpart[r] = lpart[r] * cf + ((e0 + e1) + (e2 + e3));
            corr[r]  = cf;
            p0[r] = e0; p1[r] = e1; p2[r] = e2; p3[r] = e3;
        }
#pragma unroll
        for (int r = 0; r < 8; ++r) {
            acc[0][r] *= corr[r]; acc[1][r] *= corr[r];
            acc[2][r] *= corr[r]; acc[3][r] *= corr[r];
        }

        // ---- P: D layout -> A layout via LDS, tile-major [tile][m][16] ----
        __syncthreads();
#pragma unroll
        for (int r = 0; r < 8; ++r) {
            const int m = r + hi * 8;
            Pld[0 * 256 + m * 16 + lo] = (_Float16)p0[r];
            Pld[1 * 256 + m * 16 + lo] = (_Float16)p1[r];
            Pld[2 * 256 + m * 16 + lo] = (_Float16)p2[r];
            Pld[3 * 256 + m * 16 + lo] = (_Float16)p3[r];
        }
        __syncthreads();
        const _Float16* pr = &Pld[hi * 256 + lo * 16];
        v8h a00 = *(const v8h*)(pr);          // kvlocal = hi*16 + 0..7
        v8h a01 = *(const v8h*)(pr + 8);      // kvlocal = hi*16 + 8..15
        v8h a10 = *(const v8h*)(pr + 512);    // kvlocal = 32 + hi*16 + 0..7
        v8h a11 = *(const v8h*)(pr + 520);    // kvlocal = 32 + hi*16 + 8..15
        v16h pA0, pA1;
#pragma unroll
        for (int i = 0; i < 8; ++i) {
            pA0[i] = a00[i]; pA0[i + 8] = a01[i];
            pA1[i] = a10[i]; pA1[i + 8] = a11[i];
        }

        // ---- O += P @ V over 4 DV column tiles, kv halves 0..31 / 32..63 ----
        const _Float16* vp = vtb + (size_t)lo * SEQ + kv + hi * 16;
#pragma unroll
        for (int t2 = 0; t2 < 4; ++t2) {
            v16h vb_0 = *(const v16h*)(vp + (size_t)t2 * 16 * SEQ);        // rows kv + hi*16 + 0..15
            v16h vb_1 = *(const v16h*)(vp + (size_t)t2 * 16 * SEQ + 32);   // rows kv + 32 + hi*16 + 0..15
            acc[t2] = __builtin_amdgcn_wmma_f32_16x16x32_f16(false, pA0, false, vb_0, (short)0, acc[t2], false, false);
            acc[t2] = __builtin_amdgcn_wmma_f32_16x16x32_f16(false, pA1, false, vb_1, (short)0, acc[t2], false, false);
        }
    }

    // ---- finalize: reduce l across the 16 lanes once, then normalize ----
    float* op = out + ((size_t)b * SEQ + qrow0) * DIM;
#pragma unroll
    for (int r = 0; r < 8; ++r) {
        const int m = r + hi * 8;
        const float l     = rowsum16(lpart[r]);
        const float inv_l = dropscale / l;
        op[(size_t)m * DIM + lo]      = acc[0][r] * inv_l;
        op[(size_t)m * DIM + 16 + lo] = acc[1][r] * inv_l;
        op[(size_t)m * DIM + 32 + lo] = acc[2][r] * inv_l;
        op[(size_t)m * DIM + 48 + lo] = acc[3][r] * inv_l;
    }
}

// ---------------------------------------------------------------------------
// Inputs (dict order): query, key, value, inv_scale_factor, dropout_p,
//                      Wq, bq, Wk, bk, Wv, bv
// Workspace: Qf16 (2MB) | Kf16 (2MB) | Vt f16 (2MB)
// ---------------------------------------------------------------------------
extern "C" void kernel_launch(void* const* d_in, const int* in_sizes, int n_in,
                              void* d_out, int out_size, void* d_ws, size_t ws_size,
                              hipStream_t stream) {
    const float* query = (const float*)d_in[0];
    const float* key   = (const float*)d_in[1];
    const float* value = (const float*)d_in[2];
    const float* invs  = (const float*)d_in[3];
    const float* dp    = (const float*)d_in[4];
    const float* Wq    = (const float*)d_in[5];
    const float* bq    = (const float*)d_in[6];
    const float* Wk    = (const float*)d_in[7];
    const float* bk    = (const float*)d_in[8];
    const float* Wv    = (const float*)d_in[9];
    const float* bv    = (const float*)d_in[10];

    const size_t mat = (size_t)NBATCH * SEQ * DIM;
    _Float16* qws  = (_Float16*)d_ws;
    _Float16* kws  = qws + mat;
    _Float16* vtws = kws + mat;

    dim3 pgrid(NBATCH * (SEQ / 64));
    proj_kernel<<<pgrid, 256, 0, stream>>>(query, Wq, bq, qws, 0, invs);
    proj_kernel<<<pgrid, 256, 0, stream>>>(key,   Wk, bk, kws, 0, nullptr);
    proj_kernel<<<pgrid, 256, 0, stream>>>(value, Wv, bv, vtws, 1, nullptr);

    flash_attn<<<dim3(NBATCH * (SEQ / 16)), dim3(32), 0, stream>>>(
        qws, kws, vtws, dp, (float*)d_out);
}
